// MultiScaleCrossRetention_34222299414604
// MI455X (gfx1250) — compile-verified, matrix operand order checked by reading, not run on hardware
//
#include <hip/hip_runtime.h>
#include <hip/hip_bf16.h>

#define DEVINL __device__ __forceinline__

typedef __bf16 bf16;
typedef __attribute__((ext_vector_type(16))) __bf16 v16bf;
typedef __attribute__((ext_vector_type(8)))  float  v8f;

constexpr int B_ = 2, L_ = 2048, D_ = 1024, H_ = 16, HD = 64;
constexpr float GN_EPS_F = 1e-5f;
constexpr float SCALE_BASE_F = 512.0f;
constexpr float LOG2_10000 = 13.28771237954945f;
// gammas = 1 - exp(linspace(log(1/32), log(1/512), 16))
constexpr float GAM_LO = -3.4657359027997265f;
constexpr float GAM_HI = -6.238324625039508f;

// Row-major k tile: 64 data + 8 pad = 72 bf16 = 144 B (16B-aligned rows);
// 36-dword stride -> 16 distinct banks for the 16 row-strided lanes.
constexpr int KPAD = 72;
// Transposed (K-contiguous) tiles: 32 data + 8 pad = 40 bf16 = 80 B rows
// (16B-aligned for async b128); per-lane reads are 16B-contiguous -> b128 ds
// loads instead of 16x ds_load_u16.
constexpr int VPAD = 40;

// ---- CDNA5 async global->LDS copy (ASYNCcnt) --------------------------------
DEVINL void async_b128_to_lds(uint32_t lds_byte_off, const void* gaddr) {
  asm volatile("global_load_async_to_lds_b128 %0, %1, off"
               :: "v"(lds_byte_off), "v"(gaddr)
               : "memory");
}
DEVINL void wait_async0() {
  asm volatile("s_wait_asynccnt 0" ::: "memory");
}
DEVINL uint32_t lds_off_of(const void* p) {
  return (uint32_t)(uintptr_t)p;  // low 32 bits = workgroup-relative LDS offset
}

// ---- WMMA helpers -----------------------------------------------------------
DEVINL v8f wmma_bf16(v16bf a, v16bf b, v8f c) {
  return __builtin_amdgcn_wmma_f32_16x16x32_bf16(
      false, a, false, b, (short)0, c, false, false);
}

// 16-bit A/B operand with K contiguous at p (pre-offset by row*stride+kg*8):
// element i (0..7)  <-> K = kg*8 + i        -> p[i]
// element i (8..15) <-> K = 16 + kg*8 + i-8 -> p[16 + (i-8)]
DEVINL v16bf load_op_bf16(const bf16* __restrict__ p) {
  v16bf a;
#pragma unroll
  for (int i = 0; i < 8; ++i) {
    a[i]     = p[i];
    a[8 + i] = p[16 + i];
  }
  return a;
}
DEVINL void zero_acc(v8f* acc, int n) {
  for (int j = 0; j < n; ++j)
#pragma unroll
    for (int r = 0; r < 8; ++r) acc[j][r] = 0.0f;
}

// ---- Kernel 0a: one-time f32 -> bf16 conversion -----------------------------
__global__ __launch_bounds__(256) void cvt_bf16_kernel(
    const float* __restrict__ src, bf16* __restrict__ dst, int n) {
  int i = (blockIdx.x * 256 + threadIdx.x) * 4;
  if (i + 3 < n) {
    const float4 f = *(const float4*)(src + i);
    dst[i + 0] = (__bf16)f.x;
    dst[i + 1] = (__bf16)f.y;
    dst[i + 2] = (__bf16)f.z;
    dst[i + 3] = (__bf16)f.w;
  } else {
    for (; i < n; ++i) dst[i] = (__bf16)src[i];
  }
}

// ---- Kernel 0b: one-time projection-weight transpose ------------------------
// src: (H, D, HD) f32  ->  dst: (H, HD, D) bf16  (K-contiguous B operands)
__global__ __launch_bounds__(256) void cvt_tr_w_kernel(
    const float* __restrict__ src, bf16* __restrict__ dst) {
  const int t = blockIdx.x * 256 + threadIdx.x;  // H*D*HD = 1M
  const int e = t & (HD - 1);
  const int d = (t >> 6) & (D_ - 1);
  const int h = t >> 16;
  dst[((size_t)h * HD + e) * D_ + d] = (__bf16)src[t];
}

// ---- Kernel 1: per-head projection  X(B,L,D) x WT(H,64,D) -> Out ------------
// TR_OUT=false: Out(B,H,L,64) row-major (q, k — xpos applied later)
// TR_OUT=true : Out(B,H,64,L) transposed (v — feeds K-contiguous ret-GEMM B)
template <bool TR_OUT>
__global__ __launch_bounds__(128) void proj_kernel(
    const bf16* __restrict__ X, const bf16* __restrict__ WT,
    bf16* __restrict__ Out) {
  // double-buffered 64(N=e) x 32(K=d) bf16 tile of WT, padded rows
  __shared__ __align__(16) bf16 wtile[2][HD * VPAD];

  const int tid = threadIdx.x;
  const int wave = tid >> 5, lane = tid & 31;
  const int m = lane & 15, kg = lane >> 4, hi = kg * 8;

  const int blk = blockIdx.x;           // (b*H + h)*32 + lblk
  const int lblk = blk & 31;
  const int bh = blk >> 5;
  const int b = bh >> 4, h = bh & 15;
  const int l0 = lblk * 64 + wave * 16;

  const bf16* WhT = WT + (size_t)h * HD * D_;
  const bf16* Xb = X + (size_t)b * L_ * D_;

  // stage a 64-row x 64B tile of WT (rows stride D_*2 B): 256 segs / 128 thr
  auto stage = [&](bf16* dst, const bf16* gsrc) {
#pragma unroll
    for (int p = 0; p < 2; ++p) {
      const int seg = tid + p * 128;
      const int row = seg >> 2;          // 4 x 16B segments per 64B row
      const int off = (seg & 3) * 16;
      async_b128_to_lds(lds_off_of(dst) + row * (VPAD * 2) + off,
                        (const char*)gsrc + (size_t)row * (D_ * 2) + off);
    }
  };

  v8f acc[4];
  zero_acc(acc, 4);

  stage(wtile[0], WhT);  // prologue (k0 = 0)

  for (int k0 = 0; k0 < D_; k0 += 32) {
    wait_async0();
    __syncthreads();
    const bf16* wt = wtile[(k0 >> 5) & 1];
    if (k0 + 32 < D_)
      stage(wtile[((k0 >> 5) + 1) & 1], WhT + k0 + 32);

    v16bf a = load_op_bf16(Xb + (size_t)(l0 + m) * D_ + k0 + kg * 8);
#pragma unroll
    for (int j = 0; j < 4; ++j) {
      // B column e = j*16+m; K contiguous along d in the transposed tile
      v16bf bo = load_op_bf16(wt + (j * 16 + m) * VPAD + kg * 8);
      acc[j] = wmma_bf16(a, bo, acc[j]);
    }
  }

  if (TR_OUT) {
    bf16* outp = Out + (size_t)bh * HD * L_;
#pragma unroll
    for (int j = 0; j < 4; ++j) {
      const int col = j * 16 + m;
#pragma unroll
      for (int r = 0; r < 8; ++r)  // 8 consecutive l values -> contiguous
        outp[(size_t)col * L_ + l0 + hi + r] = (__bf16)acc[j][r];
    }
  } else {
    bf16* outp = Out + (size_t)bh * L_ * HD;
#pragma unroll
    for (int j = 0; j < 4; ++j)
#pragma unroll
      for (int r = 0; r < 8; ++r)
        outp[(size_t)(l0 + r + hi) * HD + j * 16 + m] = (__bf16)acc[j][r];
  }
}

// ---- Kernel 2: xpos rotary (in place on bf16 q/k) ---------------------------
__global__ __launch_bounds__(256) void xpos_kernel(bf16* __restrict__ qp,
                                                   bf16* __restrict__ kp) {
  const int t = blockIdx.x * blockDim.x + threadIdx.x;
  const int j = t & 31;             // pair index, half = 32
  const int l = (t >> 5) & (L_ - 1);
  const int bh = t >> 16;           // b*H + h
  if (bh >= B_ * H_) return;

  const float base = (2.0f * (float)j + 0.4f * (float)HD) / (1.4f * (float)HD);
  const float p = ((float)l - (float)(L_ / 2)) / SCALE_BASE_F;
  const float sc = exp2f(p * log2f(base));          // xpos scale
  const float inv_freq = exp2f(-(float)j * (LOG2_10000 / 32.0f));
  const float ang = (float)l * inv_freq;
  float s, c;
  __sincosf(ang, &s, &c);

  const float cq = c * sc, sq = s * sc;             // downscale = False
  const float ck = c / sc, sk = s / sc;             // downscale = True

  const size_t off = ((size_t)bh * L_ + l) * HD + 2 * j;
  {
    float x1 = (float)qp[off], x2 = (float)qp[off + 1];
    qp[off]     = (__bf16)(x1 * cq - x2 * sq);
    qp[off + 1] = (__bf16)(x2 * cq + x1 * sq);
  }
  {
    float x1 = (float)kp[off], x2 = (float)kp[off + 1];
    kp[off]     = (__bf16)(x1 * ck - x2 * sk);
    kp[off + 1] = (__bf16)(x2 * ck + x1 * sk);
  }
}

// ---- Kernel 3: streaming retention + group norm -----------------------------
__global__ __launch_bounds__(128) void retention_kernel(
    const bf16* __restrict__ qx, const bf16* __restrict__ kx,
    const bf16* __restrict__ vbT, float* __restrict__ ret) {
  // double-buffered tiles shared by the 4 waves:
  //   ktile: 32 s-rows x 64 hd (row-major, row-contiguous reads)
  //   vtile: 64 e-rows x 32 s  (transposed, K-contiguous reads)
  __shared__ __align__(16) bf16 ktile[2][32 * KPAD];
  __shared__ __align__(16) bf16 vtile[2][HD * VPAD];
  __shared__ float simt[4][16][33];   // per-wave sim tile (16 x 32, padded)
  __shared__ float rtile[4][16][65];  // per-wave ret tile for group norm
  __shared__ float mu_s[4][16], rs_s[4][16];

  const int tid = threadIdx.x;
  const int wave = tid >> 5, lane = tid & 31;
  const int m = lane & 15, kg = lane >> 4, hi = kg * 8;

  const int blk = blockIdx.x;         // (b*H+h)*32 + lblk
  const int lblk = blk & 31;
  const int bh = blk >> 5;
  const int b = bh >> 4, h = bh & 15;
  const int n0 = lblk * 64 + wave * 16;

  const float gamma =
      1.0f - __expf(GAM_LO + (float)h * ((GAM_HI - GAM_LO) / 15.0f));
  const float lg2g = log2f(gamma);

  const bf16* qh = qx + (size_t)bh * L_ * HD;
  const bf16* kh = kx + (size_t)bh * L_ * HD;
  const bf16* vhT = vbT + (size_t)bh * HD * L_;

  // k tile: contiguous 4KB block (32 rows x 128B)
  auto stage_k = [&](bf16* dst, const bf16* gsrc) {
#pragma unroll
    for (int p = 0; p < 2; ++p) {
      const int seg = tid + p * 128;
      const int row = seg >> 3;
      const int off = (seg & 7) * 16;
      async_b128_to_lds(lds_off_of(dst) + row * (KPAD * 2) + off,
                        (const char*)gsrc + row * 128 + off);
    }
  };
  // v tile: 64 rows x 64B (row stride L_*2 B in global)
  auto stage_v = [&](bf16* dst, const bf16* gsrc) {
#pragma unroll
    for (int p = 0; p < 2; ++p) {
      const int seg = tid + p * 128;
      const int row = seg >> 2;
      const int off = (seg & 3) * 16;
      async_b128_to_lds(lds_off_of(dst) + row * (VPAD * 2) + off,
                        (const char*)gsrc + (size_t)row * (L_ * 2) + off);
    }
  };

  // q row operands for this wave (K = hd = 64 -> two K=32 operands)
  const v16bf aq0 = load_op_bf16(qh + (size_t)(n0 + m) * HD + kg * 8);
  const v16bf aq1 = load_op_bf16(qh + (size_t)(n0 + m) * HD + 32 + kg * 8);

  v8f acc[4];
  zero_acc(acc, 4);

  // Uniform trip count across all 4 waves (barrier safety + shared staging):
  // waves with smaller n0 get fully-masked (w==0) tail steps -> exact zeros.
  const int s_end = lblk * 64 + 64;

  stage_k(ktile[0], kh);   // prologue: s0 = 0 tiles
  stage_v(vtile[0], vhT);

  for (int s0 = 0; s0 < s_end; s0 += 32) {
    wait_async0();
    __syncthreads();  // cur tiles ready for all waves; prev-iter reads done
    const int cur = (s0 >> 5) & 1;
    const bf16* kt = ktile[cur];
    const bf16* vt = vtile[cur];
    if (s0 + 32 < s_end) {
      stage_k(ktile[cur ^ 1], kh + (size_t)(s0 + 32) * HD);
      stage_v(vtile[cur ^ 1], vhT + s0 + 32);
    }

    // --- sim = q . k^T, two 16x16 subtiles, decay-masked ---
#pragma unroll
    for (int sub = 0; sub < 2; ++sub) {
      const bf16* kr = kt + (sub * 16 + m) * KPAD + kg * 8;
      v16bf bk0 = load_op_bf16(kr);
      v16bf bk1 = load_op_bf16(kr + 32);
      v8f sim;
#pragma unroll
      for (int r = 0; r < 8; ++r) sim[r] = 0.0f;
      sim = wmma_bf16(aq0, bk0, sim);
      sim = wmma_bf16(aq1, bk1, sim);
#pragma unroll
      for (int r = 0; r < 8; ++r) {
        const int n = n0 + r + hi;
        const int s = s0 + sub * 16 + m;
        const int d = n - s;
        const float w = (d >= 0) ? exp2f((float)d * lg2g) : 0.0f;
        simt[wave][r + hi][sub * 16 + m] = sim[r] * w;
      }
    }
    __syncthreads();  // C-layout -> A-layout redistribution through LDS

    v16bf as;
#pragma unroll
    for (int i = 0; i < 8; ++i) {
      as[i]     = (__bf16)simt[wave][m][kg * 8 + i];
      as[8 + i] = (__bf16)simt[wave][m][16 + kg * 8 + i];
    }
    // --- ret += sim . v, K = 32 s-values, 4 hd subtiles ---
#pragma unroll
    for (int j = 0; j < 4; ++j) {
      // column e = j*16+m; K contiguous along s in the transposed tile
      v16bf bv = load_op_bf16(vt + (j * 16 + m) * VPAD + kg * 8);
      acc[j] = wmma_bf16(as, bv, acc[j]);
    }
    // no trailing barrier: next iteration's wait+barrier orders everything
  }

  // --- group norm over hd = 64 per row ---
#pragma unroll
  for (int j = 0; j < 4; ++j)
#pragma unroll
    for (int r = 0; r < 8; ++r) rtile[wave][r + hi][j * 16 + m] = acc[j][r];
  __syncthreads();
  if (lane < 16) {
    float sum = 0.0f, sq = 0.0f;
    for (int e = 0; e < 64; ++e) {
      float x = rtile[wave][lane][e];
      sum += x;
      sq += x * x;
    }
    const float mu = sum * (1.0f / 64.0f);
    mu_s[wave][lane] = mu;
    rs_s[wave][lane] = rsqrtf(sq * (1.0f / 64.0f) - mu * mu + GN_EPS_F);
  }
  __syncthreads();

  // write normalized, transposed to (B, L, D) with col = h*64 + e
  float* rp = ret + (size_t)b * L_ * D_ + (size_t)h * HD;
#pragma unroll
  for (int j = 0; j < 4; ++j)
#pragma unroll
    for (int r = 0; r < 8; ++r) {
      const int row = n0 + r + hi;
      const float x = (acc[j][r] - mu_s[wave][r + hi]) * rs_s[wave][r + hi];
      rp[(size_t)row * D_ + j * 16 + m] = x;
    }
}

// ---- Kernel 4: gate = silu(query @ g_w^T + g_b); gated = gate * ret (bf16) --
__global__ __launch_bounds__(128) void gate_kernel(
    const bf16* __restrict__ qbf, const bf16* __restrict__ gwb,
    const float* __restrict__ g_b, const float* __restrict__ ret,
    bf16* __restrict__ gated) {
  const int tid = threadIdx.x;
  const int wave = tid >> 5, lane = tid & 31;
  const int m = lane & 15, kg = lane >> 4, hi = kg * 8;
  const int n0 = blockIdx.x * 64 + wave * 16;  // rows over B*L = 4096
  const int j0 = blockIdx.y * 64;              // output cols

  v8f acc[4];
  zero_acc(acc, 4);

  for (int k0 = 0; k0 < D_; k0 += 32) {
    const bf16* qr = qbf + (size_t)(n0 + m) * D_ + k0 + kg * 8;
    __builtin_prefetch(qr + 64, 0, 1);  // global_prefetch_b8 (gfx1250)
    v16bf a = load_op_bf16(qr);
#pragma unroll
    for (int j = 0; j < 4; ++j) {
      const bf16* gr = gwb + (size_t)(j0 + j * 16 + m) * D_ + k0 + kg * 8;
      v16bf bo = load_op_bf16(gr);
      acc[j] = wmma_bf16(a, bo, acc[j]);
    }
  }

#pragma unroll
  for (int j = 0; j < 4; ++j)
#pragma unroll
    for (int r = 0; r < 8; ++r) {
      const int n = n0 + r + hi;
      const int col = j0 + j * 16 + m;
      const float x = acc[j][r] + g_b[col];
      const float silu = x / (1.0f + __expf(-x));
      gated[(size_t)n * D_ + col] =
          (__bf16)(silu * ret[(size_t)n * D_ + col]);
    }
}

// ---- Kernel 5: out = gated @ out_w^T + out_b (f32) --------------------------
__global__ __launch_bounds__(128) void out_kernel(
    const bf16* __restrict__ gated, const bf16* __restrict__ owb,
    const float* __restrict__ out_b, float* __restrict__ out) {
  const int tid = threadIdx.x;
  const int wave = tid >> 5, lane = tid & 31;
  const int m = lane & 15, kg = lane >> 4, hi = kg * 8;
  const int n0 = blockIdx.x * 64 + wave * 16;
  const int j0 = blockIdx.y * 64;

  v8f acc[4];
  zero_acc(acc, 4);

  for (int k0 = 0; k0 < D_; k0 += 32) {
    const bf16* ar = gated + (size_t)(n0 + m) * D_ + k0 + kg * 8;
    __builtin_prefetch(ar + 64, 0, 1);
    v16bf a = load_op_bf16(ar);
#pragma unroll
    for (int j = 0; j < 4; ++j) {
      const bf16* wr = owb + (size_t)(j0 + j * 16 + m) * D_ + k0 + kg * 8;
      v16bf bo = load_op_bf16(wr);
      acc[j] = wmma_bf16(a, bo, acc[j]);
    }
  }

#pragma unroll
  for (int j = 0; j < 4; ++j)
#pragma unroll
    for (int r = 0; r < 8; ++r) {
      const int n = n0 + r + hi;
      const int col = j0 + j * 16 + m;
      out[(size_t)n * D_ + col] = acc[j][r] + out_b[col];
    }
}

// ---- Launch -----------------------------------------------------------------
extern "C" void kernel_launch(void* const* d_in, const int* in_sizes, int n_in,
                              void* d_out, int out_size, void* d_ws,
                              size_t ws_size, hipStream_t stream) {
  const float* query = (const float*)d_in[0];
  const float* key   = (const float*)d_in[1];
  const float* value = (const float*)d_in[2];
  const float* W_Q   = (const float*)d_in[3];
  const float* W_K   = (const float*)d_in[4];
  const float* W_V   = (const float*)d_in[5];
  const float* g_w   = (const float*)d_in[6];
  const float* g_b   = (const float*)d_in[7];
  const float* out_w = (const float*)d_in[8];
  const float* out_b = (const float*)d_in[9];
  float* out = (float*)d_out;

  // workspace layout (82 MiB total; L2-resident on MI455X's 192 MB L2)
  char* ws = (char*)d_ws;
  const size_t MB = 1024u * 1024u;
  bf16*  qp    = (bf16*)(ws);             //  8 MiB: per-head q, xpos'd in place
  bf16*  kp    = (bf16*)(ws + 8 * MB);    //  8 MiB: per-head k, xpos'd in place
  bf16*  vbT   = (bf16*)(ws + 16 * MB);   //  8 MiB: per-head v, (B,H,64,L)
  float* ret   = (float*)(ws + 24 * MB);  // 16 MiB: normalized ret (B,L,D) f32
  bf16*  gated = (bf16*)(ws + 40 * MB);   //  8 MiB: silu(gate)*ret
  bf16*  wqT   = (bf16*)(ws + 48 * MB);   //  2 MiB: W_Q^T bf16 (H,64,D)
  bf16*  wkT   = (bf16*)(ws + 50 * MB);   //  2 MiB: W_K^T bf16
  bf16*  wvT   = (bf16*)(ws + 52 * MB);   //  2 MiB: W_V^T bf16
  bf16*  gwb   = (bf16*)(ws + 54 * MB);   //  2 MiB: g_w bf16
  bf16*  owb   = (bf16*)(ws + 56 * MB);   //  2 MiB: out_w bf16
  bf16*  qbf   = (bf16*)(ws + 58 * MB);   //  8 MiB: query bf16
  bf16*  kbf   = (bf16*)(ws + 66 * MB);   //  8 MiB: key bf16
  bf16*  vbf   = (bf16*)(ws + 74 * MB);   //  8 MiB: value bf16

  const dim3 blk(128);  // 4 waves (wave32); each wave owns a 16x64 WMMA tile

  // one-time operand conversions / weight transposes
  const int nW = H_ * D_ * HD;   // 1M
  const int nA = B_ * L_ * D_;   // 4M
  const int nG = D_ * D_;        // 1M
  auto cvt = [&](const float* s, bf16* d, int n) {
    cvt_bf16_kernel<<<(n / 4 + 255) / 256, 256, 0, stream>>>(s, d, n);
  };
  cvt_tr_w_kernel<<<nW / 256, 256, 0, stream>>>(W_Q, wqT);
  cvt_tr_w_kernel<<<nW / 256, 256, 0, stream>>>(W_K, wkT);
  cvt_tr_w_kernel<<<nW / 256, 256, 0, stream>>>(W_V, wvT);
  cvt(g_w, gwb, nG);
  cvt(out_w, owb, nG);
  cvt(query, qbf, nA);
  cvt(key, kbf, nA);
  cvt(value, vbf, nA);

  proj_kernel<false><<<B_ * H_ * (L_ / 64), blk, 0, stream>>>(qbf, wqT, qp);
  proj_kernel<false><<<B_ * H_ * (L_ / 64), blk, 0, stream>>>(kbf, wkT, kp);
  proj_kernel<true><<<B_ * H_ * (L_ / 64), blk, 0, stream>>>(vbf, wvT, vbT);

  xpos_kernel<<<(B_ * H_ * L_ * 32) / 256, 256, 0, stream>>>(qp, kp);

  retention_kernel<<<B_ * H_ * (L_ / 64), blk, 0, stream>>>(qp, kp, vbT, ret);

  gate_kernel<<<dim3((B_ * L_) / 64, D_ / 64), blk, 0, stream>>>(
      qbf, gwb, g_b, ret, gated);

  out_kernel<<<dim3((B_ * L_) / 64, D_ / 64), blk, 0, stream>>>(
      gated, owb, out_b, out);
}
